// MultiHeadAttention_52999896433218
// MI455X (gfx1250) — matrix operand order repaired and checked by reading, not compiled, not run on hardware
//
#include <hip/hip_runtime.h>
#include <cstdint>
#include <cstddef>

// MI455X / gfx1250, wave32. All GEMMs via v_wmma_f32_16x16x32_bf16.
// B*F = 64 batches, S = 512, D = 512, H = 8, DK = DV = 64.
// Each wave computes a 32x64 output tile: 2 A tiles x 4 B tiles = 8 WMMA/k-step.

#define NBF 64
#define SEQ 512
#define DM  512
#define NH  8
#define HD  64

typedef __bf16 v16bf __attribute__((ext_vector_type(16)));
typedef float  v8f   __attribute__((ext_vector_type(8)));

union BF16Tile {
    v16bf  v;
    uint4  q[2];
    __bf16 h[16];
};

__device__ __forceinline__ v8f wmma_bf16(const BF16Tile& a, const BF16Tile& b, v8f c) {
    return __builtin_amdgcn_wmma_f32_16x16x32_bf16(
        /*neg_a=*/false, a.v, /*neg_b=*/false, b.v,
        /*c_mod=*/(short)0, c, /*reuse_a=*/false, /*reuse_b=*/false);
}

__device__ __forceinline__ void cvt8(float4 f0, float4 f1, __bf16* h) {
    h[0] = (__bf16)f0.x; h[1] = (__bf16)f0.y; h[2] = (__bf16)f0.z; h[3] = (__bf16)f0.w;
    h[4] = (__bf16)f1.x; h[5] = (__bf16)f1.y; h[6] = (__bf16)f1.z; h[7] = (__bf16)f1.w;
}

__device__ __forceinline__ void cvt8exp(float4 f0, float4 f1, float mx, float inv, __bf16* h) {
    h[0] = (__bf16)(__expf(f0.x - mx) * inv); h[1] = (__bf16)(__expf(f0.y - mx) * inv);
    h[2] = (__bf16)(__expf(f0.z - mx) * inv); h[3] = (__bf16)(__expf(f0.w - mx) * inv);
    h[4] = (__bf16)(__expf(f1.x - mx) * inv); h[5] = (__bf16)(__expf(f1.y - mx) * inv);
    h[6] = (__bf16)(__expf(f1.z - mx) * inv); h[7] = (__bf16)(__expf(f1.w - mx) * inv);
}

// ---------------------------------------------------------------------------
// K0: W [k][n] fp32 -> Wt [n][k] bf16  (coalesced writes). 512x512 per weight.
// ---------------------------------------------------------------------------
__global__ void cvtw_kernel(const float* __restrict__ W, __bf16* __restrict__ Wt) {
    int i = blockIdx.x * blockDim.x + threadIdx.x;   // 0..262143
    int n = i >> 9, k = i & 511;
    Wt[i] = (__bf16)W[(size_t)k * DM + n];
}

// ---------------------------------------------------------------------------
// K1: projections. P = X @ W for Q/K/V; head-split bf16 stores.
//   Q,K -> [bf*H + h][s][dk] (row-major over dk)
//   V   -> [bf*H + h][dv][t] (transposed: contiguous t for WMMA B loads)
// mat selector comes from blockIdx.y so it is compiler-provably uniform:
// store-path selection lowers to scalar branches, not exec-mask predication.
// ---------------------------------------------------------------------------
__global__ void proj_kernel(const float* __restrict__ xq, const float* __restrict__ xk,
                            const float* __restrict__ xv,
                            const __bf16* __restrict__ wqt,
                            const __bf16* __restrict__ wkt,
                            const __bf16* __restrict__ wvt,
                            __bf16* __restrict__ qh,
                            __bf16* __restrict__ kh,
                            __bf16* __restrict__ vt) {
    int wid = threadIdx.x >> 5, lane = threadIdx.x & 31;
    int mat = blockIdx.y;                  // 0=Q 1=K 2=V (uniform, scalar)
    int g = blockIdx.x * 8 + wid;          // 0..8191
    int bf  = g >> 7;
    int tile = g & 127;
    int s0 = (tile >> 3) << 5;
    int n0 = (tile & 7) << 6;
    int r = lane & 15, hi = lane >> 4;

    const float* X = (mat == 0 ? xq : (mat == 1 ? xk : xv)) + (size_t)bf * SEQ * DM;
    const __bf16* Wt = (mat == 0 ? wqt : (mat == 1 ? wkt : wvt));

    const float* arow0 = X + (size_t)(s0 + r) * DM;
    const float* arow1 = X + (size_t)(s0 + 16 + r) * DM;
    v8f acc[2][4] = {};

    for (int kk = 0; kk < 16; ++kk) {
        int kb = kk * 32;
        BF16Tile a[2];
        {
            const float4* p0 = (const float4*)(arow0 + kb + 8 * hi);
            const float4* p1 = (const float4*)(arow0 + kb + 16 + 8 * hi);
            cvt8(p0[0], p0[1], &a[0].h[0]);
            cvt8(p1[0], p1[1], &a[0].h[8]);
            const float4* p2 = (const float4*)(arow1 + kb + 8 * hi);
            const float4* p3 = (const float4*)(arow1 + kb + 16 + 8 * hi);
            cvt8(p2[0], p2[1], &a[1].h[0]);
            cvt8(p3[0], p3[1], &a[1].h[8]);
        }
        BF16Tile b[4];
#pragma unroll
        for (int t = 0; t < 4; ++t) {
            const uint4* pb = (const uint4*)(Wt + (size_t)(n0 + t * 16 + r) * DM + kb + 8 * hi);
            b[t].q[0] = pb[0]; b[t].q[1] = pb[2];
        }
#pragma unroll
        for (int t = 0; t < 4; ++t) {
            acc[0][t] = wmma_bf16(a[0], b[t], acc[0][t]);
            acc[1][t] = wmma_bf16(a[1], b[t], acc[1][t]);
        }
    }

    if (mat == 2) {
#pragma unroll
        for (int mi = 0; mi < 2; ++mi)
#pragma unroll
            for (int t = 0; t < 4; ++t)
#pragma unroll
                for (int rr = 0; rr < 8; ++rr) {
                    int m = 16 * mi + rr + 8 * hi;
                    int n = n0 + t * 16 + r;
                    vt[((size_t)(bf * NH + (n >> 6)) * HD + (n & 63)) * SEQ + s0 + m] =
                        (__bf16)acc[mi][t][rr];
                }
    } else {
        __bf16* dst = (mat == 0) ? qh : kh;
#pragma unroll
        for (int mi = 0; mi < 2; ++mi)
#pragma unroll
            for (int t = 0; t < 4; ++t)
#pragma unroll
                for (int rr = 0; rr < 8; ++rr) {
                    int m = 16 * mi + rr + 8 * hi;
                    int n = n0 + t * 16 + r;
                    dst[((size_t)(bf * NH + (n >> 6)) * SEQ + s0 + m) * HD + (n & 63)] =
                        (__bf16)acc[mi][t][rr];
                }
    }
}

// ---------------------------------------------------------------------------
// K2: scores[s][t] = (Q @ K^T) * 1/sqrt(64) + res_att, fp32 -> d_out.
// res_att read + scores store are non-temporal (pure HBM streams; keep L2 for
// the bf16 operand tensors which fit in the 192MB L2).
// ---------------------------------------------------------------------------
__global__ void scores_kernel(const __bf16* __restrict__ qh,
                              const __bf16* __restrict__ kh,
                              const float* __restrict__ res,
                              float* __restrict__ sc) {
    int wid = threadIdx.x >> 5, lane = threadIdx.x & 31;
    int g = blockIdx.x * 8 + wid;          // 0..65535
    int bfh = g >> 7;
    int tile = g & 127;
    int s0 = (tile >> 3) << 5;
    int t0 = (tile & 7) << 6;
    int r = lane & 15, hi = lane >> 4;

    const __bf16* arow0 = qh + ((size_t)bfh * SEQ + s0 + r) * HD;
    const __bf16* arow1 = qh + ((size_t)bfh * SEQ + s0 + 16 + r) * HD;
    const __bf16* kb0   = kh + (size_t)bfh * SEQ * HD;
    v8f acc[2][4] = {};

#pragma unroll
    for (int kk = 0; kk < 2; ++kk) {
        int kb = kk * 32;
        BF16Tile a[2];
        a[0].q[0] = *(const uint4*)(arow0 + kb + 8 * hi);
        a[0].q[1] = *(const uint4*)(arow0 + kb + 16 + 8 * hi);
        a[1].q[0] = *(const uint4*)(arow1 + kb + 8 * hi);
        a[1].q[1] = *(const uint4*)(arow1 + kb + 16 + 8 * hi);
        BF16Tile b[4];
#pragma unroll
        for (int t = 0; t < 4; ++t) {
            const uint4* pb = (const uint4*)(kb0 + (size_t)(t0 + t * 16 + r) * HD + kb + 8 * hi);
            b[t].q[0] = pb[0]; b[t].q[1] = pb[2];
        }
#pragma unroll
        for (int t = 0; t < 4; ++t) {
            acc[0][t] = wmma_bf16(a[0], b[t], acc[0][t]);
            acc[1][t] = wmma_bf16(a[1], b[t], acc[1][t]);
        }
    }

    const float scale = 0.125f;            // 1/sqrt(64)
#pragma unroll
    for (int mi = 0; mi < 2; ++mi) {
#pragma unroll
        for (int t = 0; t < 4; ++t) {
#pragma unroll
            for (int rr = 0; rr < 8; ++rr) {
                int m = 16 * mi + rr + 8 * hi;
                size_t idx = ((size_t)bfh * SEQ + s0 + m) * SEQ + t0 + t * 16 + r;
                float rv = __builtin_nontemporal_load(&res[idx]);
                __builtin_nontemporal_store(acc[mi][t][rr] * scale + rv, &sc[idx]);
            }
        }
    }
}

// ---------------------------------------------------------------------------
// K3: fused softmax + context. Per wave: one (bf,h, 32-row) stripe.
// Stats: lane l owns full row s0+l (512 elems) -> no reduction needed; stats
// redistributed to the WMMA A row mapping with two __shfl's. A tiles are built
// by applying exp(x-max)*inv on the fly (attn probs never touch HBM).
// ---------------------------------------------------------------------------
__global__ void attnctx_kernel(const float* __restrict__ sc,
                               const __bf16* __restrict__ vt,
                               __bf16* __restrict__ ctx) {
    int wid = threadIdx.x >> 5, lane = threadIdx.x & 31;
    int g = blockIdx.x * 8 + wid;          // 0..8191
    int bfh = g >> 4;
    int s0 = (g & 15) << 5;
    int r = lane & 15, hi = lane >> 4;

    // --- per-lane row stats (row s0 + lane) ---
    const float4* prow = (const float4*)(sc + ((size_t)bfh * SEQ + s0 + lane) * SEQ);
    float mx = -3.402823466e38f;
    for (int i = 0; i < 128; ++i) {
        float4 f = prow[i];
        mx = fmaxf(mx, fmaxf(fmaxf(f.x, f.y), fmaxf(f.z, f.w)));
    }
    float sum = 0.f;
    for (int i = 0; i < 128; ++i) {
        float4 f = prow[i];
        sum += __expf(f.x - mx) + __expf(f.y - mx) + __expf(f.z - mx) + __expf(f.w - mx);
    }
    float inv = 1.0f / sum;

    // redistribute stats: A tile mi needs stats of row (16*mi + r)
    float mx0  = __shfl(mx,  r, 32),      inv0 = __shfl(inv, r, 32);
    float mx1  = __shfl(mx,  16 + r, 32), inv1 = __shfl(inv, 16 + r, 32);

    const float* arow0 = sc + ((size_t)bfh * SEQ + s0 + r) * SEQ;
    const float* arow1 = sc + ((size_t)bfh * SEQ + s0 + 16 + r) * SEQ;
    const __bf16* vb = vt + (size_t)bfh * HD * SEQ;
    v8f acc[2][4] = {};

    for (int kk = 0; kk < 16; ++kk) {
        int kb = kk * 32;
        BF16Tile a[2];
        {
            const float4* p0 = (const float4*)(arow0 + kb + 8 * hi);
            const float4* p1 = (const float4*)(arow0 + kb + 16 + 8 * hi);
            cvt8exp(p0[0], p0[1], mx0, inv0, &a[0].h[0]);
            cvt8exp(p1[0], p1[1], mx0, inv0, &a[0].h[8]);
            const float4* p2 = (const float4*)(arow1 + kb + 8 * hi);
            const float4* p3 = (const float4*)(arow1 + kb + 16 + 8 * hi);
            cvt8exp(p2[0], p2[1], mx1, inv1, &a[1].h[0]);
            cvt8exp(p3[0], p3[1], mx1, inv1, &a[1].h[8]);
        }
        BF16Tile b[4];
#pragma unroll
        for (int t = 0; t < 4; ++t) {
            const uint4* pb = (const uint4*)(vb + (size_t)(t * 16 + r) * SEQ + kb + 8 * hi);
            b[t].q[0] = pb[0]; b[t].q[1] = pb[2];
        }
#pragma unroll
        for (int t = 0; t < 4; ++t) {
            acc[0][t] = wmma_bf16(a[0], b[t], acc[0][t]);
            acc[1][t] = wmma_bf16(a[1], b[t], acc[1][t]);
        }
    }

    int bf = bfh >> 3, h = bfh & 7;
#pragma unroll
    for (int mi = 0; mi < 2; ++mi) {
#pragma unroll
        for (int t = 0; t < 4; ++t) {
#pragma unroll
            for (int rr = 0; rr < 8; ++rr) {
                int m = 16 * mi + rr + 8 * hi;
                ctx[((size_t)bf * SEQ + s0 + m) * DM + h * HD + t * 16 + r] =
                    (__bf16)acc[mi][t][rr];
            }
        }
    }
}

// ---------------------------------------------------------------------------
// K4: output = tanh(Ctx @ W_fc + b_fc) + input_Q, fp32 -> d_out (NT store).
// ---------------------------------------------------------------------------
__global__ void fc_kernel(const __bf16* __restrict__ ctx,
                          const __bf16* __restrict__ wfct,
                          const float* __restrict__ bfc,
                          const float* __restrict__ inq,
                          float* __restrict__ out) {
    int wid = threadIdx.x >> 5, lane = threadIdx.x & 31;
    int g = blockIdx.x * 8 + wid;          // 0..8191
    int bf = g >> 7;
    int tile = g & 127;
    int s0 = (tile >> 3) << 5;
    int n0 = (tile & 7) << 6;
    int r = lane & 15, hi = lane >> 4;

    const __bf16* arow0 = ctx + ((size_t)bf * SEQ + s0 + r) * DM;
    const __bf16* arow1 = ctx + ((size_t)bf * SEQ + s0 + 16 + r) * DM;
    v8f acc[2][4] = {};

    for (int kk = 0; kk < 16; ++kk) {
        int kb = kk * 32;
        BF16Tile a[2];
        a[0].q[0] = *(const uint4*)(arow0 + kb + 8 * hi);
        a[0].q[1] = *(const uint4*)(arow0 + kb + 16 + 8 * hi);
        a[1].q[0] = *(const uint4*)(arow1 + kb + 8 * hi);
        a[1].q[1] = *(const uint4*)(arow1 + kb + 16 + 8 * hi);
        BF16Tile b[4];
#pragma unroll
        for (int t = 0; t < 4; ++t) {
            const uint4* pb = (const uint4*)(wfct + (size_t)(n0 + t * 16 + r) * DM + kb + 8 * hi);
            b[t].q[0] = pb[0]; b[t].q[1] = pb[2];
        }
#pragma unroll
        for (int t = 0; t < 4; ++t) {
            acc[0][t] = wmma_bf16(a[0], b[t], acc[0][t]);
            acc[1][t] = wmma_bf16(a[1], b[t], acc[1][t]);
        }
    }

#pragma unroll
    for (int mi = 0; mi < 2; ++mi) {
#pragma unroll
        for (int t = 0; t < 4; ++t) {
#pragma unroll
            for (int rr = 0; rr < 8; ++rr) {
                int m = 16 * mi + rr + 8 * hi;
                int n = n0 + t * 16 + r;
                size_t idx = ((size_t)bf * SEQ + s0 + m) * DM + n;
                float v = tanhf(acc[mi][t][rr] + bfc[n]) + inq[idx];
                __builtin_nontemporal_store(v, &out[idx]);
            }
        }
    }
}

// ---------------------------------------------------------------------------
// Host-side launch. Workspace (bf16 elements, ~136 MB total):
//   WqT WkT WvT WfcT : 4 x 262144
//   Qh  Kh  Vt  Ctx  : 4 x 16777216
// d_out: [output 16,777,216 f32][scores 134,217,728 f32]
// ---------------------------------------------------------------------------
extern "C" void kernel_launch(void* const* d_in, const int* in_sizes, int n_in,
                              void* d_out, int out_size, void* d_ws, size_t ws_size,
                              hipStream_t stream) {
    const float* inQ = (const float*)d_in[0];
    const float* inK = (const float*)d_in[1];
    const float* inV = (const float*)d_in[2];
    const float* res = (const float*)d_in[3];
    const float* WQ  = (const float*)d_in[4];
    const float* WK  = (const float*)d_in[5];
    const float* WV  = (const float*)d_in[6];
    const float* Wfc = (const float*)d_in[7];
    const float* bfc = (const float*)d_in[8];

    float* out = (float*)d_out;
    float* sc  = out + (size_t)NBF * SEQ * DM;     // scores region

    __bf16* wqt  = (__bf16*)d_ws;
    __bf16* wkt  = wqt  + 262144;
    __bf16* wvt  = wkt  + 262144;
    __bf16* wfct = wvt  + 262144;
    __bf16* qh   = wfct + 262144;
    __bf16* kh   = qh   + (size_t)16777216;
    __bf16* vt   = kh   + (size_t)16777216;
    __bf16* ctx  = vt   + (size_t)16777216;

    cvtw_kernel<<<1024, 256, 0, stream>>>(WQ,  wqt);
    cvtw_kernel<<<1024, 256, 0, stream>>>(WK,  wkt);
    cvtw_kernel<<<1024, 256, 0, stream>>>(WV,  wvt);
    cvtw_kernel<<<1024, 256, 0, stream>>>(Wfc, wfct);

    proj_kernel   <<<dim3(1024, 3), 256, 0, stream>>>(inQ, inK, inV, wqt, wkt, wvt, qh, kh, vt);
    scores_kernel <<<8192, 256, 0, stream>>>(qh, kh, res, sc);
    attnctx_kernel<<<1024, 256, 0, stream>>>(sc, vt, ctx);
    fc_kernel     <<<1024, 256, 0, stream>>>(ctx, wfct, bfc, inQ, out);
}
